// AttentionMechanism_15436112461862
// MI455X (gfx1250) — matrix operand order repaired and checked by reading, not compiled, not run on hardware
//
#include <hip/hip_runtime.h>
#include <hip/hip_bf16.h>

#define H 512
#define B 64
#define S 2048
#define LDA 520   // padded bf16 row stride for the LDS A tile (bank-conflict avoidance)

typedef __bf16 bf16_t;
typedef bf16_t v16bf __attribute__((ext_vector_type(16)));
typedef bf16_t v8bf  __attribute__((ext_vector_type(8)));
typedef float  v8f   __attribute__((ext_vector_type(8)));

#if __has_builtin(__builtin_amdgcn_global_load_async_to_lds_b128)
#define HAVE_ASYNC_LDS 1
typedef int v4i __attribute__((vector_size(16)));
typedef __attribute__((address_space(1))) v4i* gv4i_t;
typedef __attribute__((address_space(3))) v4i* lv4i_t;
#endif

__device__ __forceinline__ void wait_asynccnt0() {
#if __has_builtin(__builtin_amdgcn_s_wait_asynccnt)
    __builtin_amdgcn_s_wait_asynccnt(0);
#else
    asm volatile("s_wait_asynccnt 0x0" ::: "memory");
#endif
}

// ---------------- K0: W_enc -> bf16 in WMMA B-fragment-contiguous layout ----------------
// Wk element order: [kt(16)][nt(32)][lane(32)][e(16)]  (bf16)
// lane l = half*16 + row holds, for tile (kt,nt): column n = nt*16 + row,
// K elements kt*32 + half*16 + e  (e = 0..15)  -> contiguous k order, matching the A fragment reads.
__global__ void prep_wenc_kernel(const float* __restrict__ W_attn, bf16_t* __restrict__ Wk) {
    int idx = blockIdx.x * blockDim.x + threadIdx.x;   // 0 .. 512*512-1
    int e  = idx & 15;
    int l  = (idx >> 4) & 31;
    int nt = (idx >> 9) & 31;
    int kt = (idx >> 14) & 15;
    int k  = kt * 32 + (l >> 4) * 16 + e;
    int n  = nt * 16 + (l & 15);
    Wk[idx] = (bf16_t)W_attn[(H + k) * H + n];
}

// ---------------- K1: dec_proj = decoder_hidden @ W_dec + b_attn  (tiny) ----------------
__global__ void dec_proj_kernel(const float* __restrict__ dh, const float* __restrict__ W_attn,
                                const float* __restrict__ b_attn, float* __restrict__ dec_proj) {
    int h = threadIdx.x;   // 0..511
    int b = blockIdx.x;    // 0..63
    float sum = b_attn[h];
    const float* dhb = dh + b * H;
    for (int k = 0; k < H; ++k) sum = fmaf(dhb[k], W_attn[k * H + h], sum);
    dec_proj[b * H + h] = sum;
}

// ---------------- K2: fused enc_proj GEMM (bf16 WMMA) + tanh + dot(v) -> scores ----------------
// grid: (S/64, B), block: 512 (16 waves). Each WG: 64 S-rows x 512 cols, K=512.
// B slices (32 KB) are double-buffered through LDS via gfx1250 async global->LDS DMA
// (ASYNCcnt); the wave only waits on asynccnt at the end of the WMMA block.
__global__ __launch_bounds__(512)
void scores_kernel(const float* __restrict__ enc,      // [B,S,H] fp32
                   const bf16_t* __restrict__ Wk,      // [16][32][32][16] bf16 (frag layout)
                   const float* __restrict__ dec_proj, // [B,H]
                   const float* __restrict__ v,        // [H]
                   float* __restrict__ scores) {       // [B,S]
    __shared__ __attribute__((aligned(32))) bf16_t ldsA[64 * LDA];       // 66560 B
    __shared__ __attribute__((aligned(32))) bf16_t ldsB[2][512 * 32];    // 2 x 32 KB
    __shared__ float dec_lds[H];
    __shared__ float v_lds[H];
    __shared__ float sc_lds[64];

    const int tid = threadIdx.x;
    const int b   = blockIdx.y;
    const int s0  = blockIdx.x * 64;

    for (int i = tid; i < H; i += 512) {
        dec_lds[i] = dec_proj[b * H + i];
        v_lds[i]   = v[i];
    }
    if (tid < 64) sc_lds[tid] = 0.0f;

    // Kick off async staging of B slice kt=0 (contiguous: thread t moves 64 B)
#ifdef HAVE_ASYNC_LDS
    {
        const bf16_t* src = Wk + (size_t)tid * 32;
        bf16_t* dst = &ldsB[0][tid * 32];
        gv4i_t gp = (gv4i_t)(uintptr_t)src;   // global AS(1) == flat VA for global memory
        lv4i_t lp = (lv4i_t)dst;              // addrspacecast generic -> LDS
        __builtin_amdgcn_global_load_async_to_lds_b128(gp, lp, 0, 0);
        __builtin_amdgcn_global_load_async_to_lds_b128(gp, lp, 16, 0);
        __builtin_amdgcn_global_load_async_to_lds_b128(gp, lp, 32, 0);
        __builtin_amdgcn_global_load_async_to_lds_b128(gp, lp, 48, 0);
    }
#endif

    // Stage enc tile 64x512 fp32 -> bf16 in LDS (A tile, read once, used by all 16 kt steps)
    const float* encb = enc + ((size_t)b * S + s0) * H;
    for (int it = 0; it < 16; ++it) {
        int flat = (it * 512 + tid) * 4;
        int r = flat >> 9;
        int c = flat & 511;
        const float4 f = *(const float4*)(encb + (size_t)r * H + c);
        bf16_t* dst = &ldsA[r * LDA + c];
        dst[0] = (bf16_t)f.x; dst[1] = (bf16_t)f.y;
        dst[2] = (bf16_t)f.z; dst[3] = (bf16_t)f.w;
    }

#ifndef HAVE_ASYNC_LDS
    {   // fallback: stage B slice 0 through registers
        const bf16_t* src = Wk + (size_t)tid * 32;
        v16bf r0 = *(const v16bf*)(src);
        v16bf r1 = *(const v16bf*)(src + 16);
        *(v16bf*)&ldsB[0][tid * 32]      = r0;
        *(v16bf*)&ldsB[0][tid * 32 + 16] = r1;
    }
#else
    wait_asynccnt0();
#endif
    __syncthreads();

    const int lane = tid & 31;
    const int wave = tid >> 5;       // 0..15
    const int row  = lane & 15;      // M index (A/C) and N index (B/C)
    const int half = lane >> 4;
    const int mt   = wave & 3;       // M tile: rows mt*16 .. mt*16+15
    const int ng   = wave >> 2;      // N group: tiles ng*8 .. ng*8+7

    v8f acc[8];
    const v8f vzero = {0.f, 0.f, 0.f, 0.f, 0.f, 0.f, 0.f, 0.f};
#pragma unroll
    for (int j = 0; j < 8; ++j) acc[j] = vzero;

    for (int kt = 0; kt < 16; ++kt) {
        const int cur = kt & 1;

#ifdef HAVE_ASYNC_LDS
        // Launch async DMA of the next B slice; completes while we do WMMAs.
        if (kt < 15) {
            const bf16_t* src = Wk + ((size_t)(kt + 1) * 512 + tid) * 32;
            bf16_t* dst = &ldsB[1 - cur][tid * 32];
            gv4i_t gp = (gv4i_t)(uintptr_t)src;
            lv4i_t lp = (lv4i_t)dst;
            __builtin_amdgcn_global_load_async_to_lds_b128(gp, lp, 0, 0);
            __builtin_amdgcn_global_load_async_to_lds_b128(gp, lp, 16, 0);
            __builtin_amdgcn_global_load_async_to_lds_b128(gp, lp, 32, 0);
            __builtin_amdgcn_global_load_async_to_lds_b128(gp, lp, 48, 0);
        }
#else
        v16bf p0, p1;
        if (kt < 15) {
            const bf16_t* src = Wk + ((size_t)(kt + 1) * 512 + tid) * 32;
            p0 = *(const v16bf*)(src);
            p1 = *(const v16bf*)(src + 16);
        }
#endif

        // A fragment (16x32 bf16) from LDS
        const bf16_t* ap = &ldsA[(mt * 16 + row) * LDA + kt * 32 + half * 16];
        v16bf a;
        ((v8bf*)&a)[0] = *(const v8bf*)(ap);
        ((v8bf*)&a)[1] = *(const v8bf*)(ap + 8);

#pragma unroll
        for (int j = 0; j < 8; ++j) {
            const int nt = ng * 8 + j;
            // fragment-contiguous: 32 B per lane, lane-consecutive -> conflict-free
            const v16bf bb = *(const v16bf*)&ldsB[cur][(nt * 32 + lane) * 16];
            acc[j] = __builtin_amdgcn_wmma_f32_16x16x32_bf16(
                false, a, false, bb, (short)0, acc[j], false, false);
        }

#ifdef HAVE_ASYNC_LDS
        if (kt < 15) wait_asynccnt0();   // DMA had the whole WMMA block to finish
#else
        if (kt < 15) {
            *(v16bf*)&ldsB[1 - cur][tid * 32]      = p0;
            *(v16bf*)&ldsB[1 - cur][tid * 32 + 16] = p1;
        }
#endif
        __syncthreads();
    }

    // Epilogue: e = tanh(acc + dec_proj[col]); partial score = sum_col e * v[col]
#pragma unroll
    for (int j = 0; j < 8; ++j) {
        const int col = (ng * 8 + j) * 16 + row;
        const float dp = dec_lds[col];
        const float vv = v_lds[col];
#pragma unroll
        for (int r = 0; r < 8; ++r) {
            // C layout: VGPR r, lanes 0-15 => M = r, lanes 16-31 => M = r+8; N = row
            float e = tanhf(acc[j][r] + dp) * vv;
            e += __shfl_xor(e, 1);
            e += __shfl_xor(e, 2);
            e += __shfl_xor(e, 4);
            e += __shfl_xor(e, 8);
            if (row == 0) atomicAdd(&sc_lds[mt * 16 + half * 8 + r], e);  // ds_add_f32
        }
    }
    __syncthreads();
    if (tid < 64) scores[(size_t)b * S + s0 + tid] = sc_lds[tid];
}

// ---------------- K3: softmax over S per batch ----------------
__global__ __launch_bounds__(256)
void softmax_kernel(const float* __restrict__ scores, float* __restrict__ weights) {
    const int b = blockIdx.x;
    const int tid = threadIdx.x;
    __shared__ float redm[8];
    __shared__ float reds[8];
    const float* sr = scores + (size_t)b * S;

    float x[8];
    float m = -1e30f;
#pragma unroll
    for (int i = 0; i < 8; ++i) { x[i] = sr[tid + 256 * i]; m = fmaxf(m, x[i]); }
    for (int off = 16; off; off >>= 1) m = fmaxf(m, __shfl_xor(m, off));
    if ((tid & 31) == 0) redm[tid >> 5] = m;
    __syncthreads();
    m = redm[0];
#pragma unroll
    for (int i = 1; i < 8; ++i) m = fmaxf(m, redm[i]);

    float sum = 0.0f;
#pragma unroll
    for (int i = 0; i < 8; ++i) { x[i] = __expf(x[i] - m); sum += x[i]; }
    for (int off = 16; off; off >>= 1) sum += __shfl_xor(sum, off);
    if ((tid & 31) == 0) reds[tid >> 5] = sum;
    __syncthreads();
    float tot = 0.0f;
#pragma unroll
    for (int i = 0; i < 8; ++i) tot += reds[i];
    const float inv = 1.0f / tot;

    float* wr = weights + (size_t)b * S;
#pragma unroll
    for (int i = 0; i < 8; ++i) wr[tid + 256 * i] = x[i] * inv;
}

// ---------------- K4: context = weights @ encoder_outputs (bandwidth pass) ----------------
__global__ __launch_bounds__(256)
void context_kernel(const float* __restrict__ enc, const float* __restrict__ weights,
                    float* __restrict__ ctx) {
    const int b = blockIdx.y;
    const int h = blockIdx.x * 256 + threadIdx.x;
    const float* encb = enc + (size_t)b * S * H;
    const float* wb   = weights + (size_t)b * S;
    float sum = 0.0f;
    for (int s = 0; s < S; ++s) sum = fmaf(wb[s], encb[(size_t)s * H + h], sum);
    ctx[b * H + h] = sum;
}

extern "C" void kernel_launch(void* const* d_in, const int* in_sizes, int n_in,
                              void* d_out, int out_size, void* d_ws, size_t ws_size,
                              hipStream_t stream) {
    (void)in_sizes; (void)n_in; (void)out_size; (void)ws_size;
    const float* dh   = (const float*)d_in[0];  // decoder_hidden [B,H]
    const float* enc  = (const float*)d_in[1];  // encoder_outputs [B,S,H]
    const float* W    = (const float*)d_in[2];  // W_attn [2H,H]
    const float* bias = (const float*)d_in[3];  // b_attn [H]
    const float* v    = (const float*)d_in[4];  // v [H]

    float* out     = (float*)d_out;
    float* ctx     = out;            // [B,H]   = 32768 floats
    float* weights = out + B * H;    // [B,S]   = 131072 floats

    char* ws = (char*)d_ws;
    bf16_t* Wk      = (bf16_t*)ws;                                  // 512 KB (frag layout)
    float* dec_proj = (float*)(ws + 512 * 1024);                    // 128 KB
    float* scores   = (float*)(ws + 512 * 1024 + 128 * 1024);       // 512 KB

    prep_wenc_kernel<<<(H * H) / 256, 256, 0, stream>>>(W, Wk);
    dec_proj_kernel<<<B, H, 0, stream>>>(dh, W, bias, dec_proj);
    scores_kernel<<<dim3(S / 64, B), 512, 0, stream>>>(enc, Wk, dec_proj, v, scores);
    softmax_kernel<<<B, 256, 0, stream>>>(scores, weights);
    context_kernel<<<dim3(H / 256, B), 256, 0, stream>>>(enc, weights, ctx);
}
